// MultiHeadAttention_45509473469053
// MI455X (gfx1250) — matrix operand order repaired and checked by reading, not compiled, not run on hardware
//
#include <hip/hip_runtime.h>
#include <math.h>
#include <stdint.h>

// ---------------- problem constants ----------------
static constexpr int B_   = 2;
static constexpr int L_   = 1920;
static constexpr int DM_  = 512;
static constexpr int H_   = 8;
static constexpr int DEP_ = 64;                 // DM_/H_
static constexpr int BH_  = B_ * H_;            // 16
static constexpr int M_   = B_ * L_;            // 3840 projection rows
static constexpr int LT_  = L_ / 16;            // 120 key tiles

typedef __attribute__((ext_vector_type(16))) _Float16 v16h;
typedef __attribute__((ext_vector_type(8)))  _Float16 h8;
typedef __attribute__((ext_vector_type(8)))  float    v8f;
typedef __attribute__((ext_vector_type(4)))  float    f4;

// ---------------- WMMA helper (gfx1250, wave32) ----------------
__device__ __forceinline__ v8f wmma_f16(v16h a, v16h b, v8f c) {
  return __builtin_amdgcn_wmma_f32_16x16x32_f16(false, a, false, b, (short)0, c,
                                                false, false);
}

// ---------------- fragment builders -----------------------------
// A fragment 16x32 (MxK): lane holds row M=lane&15; slot s -> K = kb+s (s<8)
// or kb+8+s (s>=8), kb = 8*(lane>=16).  All loads 16B-aligned vectors.
__device__ __forceinline__ v16h a_frag_f32(const float* __restrict__ p, int ld, int lane) {
  const float* row = p + (lane & 15) * ld + ((lane & 16) >> 1);
  f4 a0 = *(const f4*)(row);
  f4 a1 = *(const f4*)(row + 4);
  f4 b0 = *(const f4*)(row + 16);
  f4 b1 = *(const f4*)(row + 20);
  v16h r;
#pragma unroll
  for (int s = 0; s < 4; ++s) {
    r[s]      = (_Float16)a0[s];
    r[s + 4]  = (_Float16)a1[s];
    r[s + 8]  = (_Float16)b0[s];
    r[s + 12] = (_Float16)b1[s];
  }
  return r;
}

__device__ __forceinline__ v16h a_frag_f16(const _Float16* __restrict__ p, int ld, int lane) {
  const _Float16* row = p + (lane & 15) * ld + ((lane & 16) >> 1);
  h8 lo = *(const h8*)(row);
  h8 hi = *(const h8*)(row + 16);
  v16h r;
#pragma unroll
  for (int s = 0; s < 8; ++s) { r[s] = lo[s]; r[s + 8] = hi[s]; }
  return r;
}

// A fragment from the LDS f32 probability strip -> ds_load_b128 traffic
__device__ __forceinline__ v16h a_frag_lds(const float* sm, int ld, int c0, int lane) {
  const float* row = sm + (lane & 15) * ld + c0 + ((lane & 16) >> 1);
  f4 a0 = *(const f4*)(row);
  f4 a1 = *(const f4*)(row + 4);
  f4 b0 = *(const f4*)(row + 16);
  f4 b1 = *(const f4*)(row + 20);
  v16h r;
#pragma unroll
  for (int s = 0; s < 4; ++s) {
    r[s]      = (_Float16)a0[s];
    r[s + 4]  = (_Float16)a1[s];
    r[s + 8]  = (_Float16)b0[s];
    r[s + 12] = (_Float16)b1[s];
  }
  return r;
}

// B fragment 32x16 (KxN) stored N-major: element(K,N)=p[N*ld+K].
// Lane: N=lane&15, K = kb+s (s<8) or kb+8+s (s>=8), kb = 8*(lane>=16) -> two 16B vector loads.
__device__ __forceinline__ v16h b_frag_T_f16(const _Float16* __restrict__ p, int ld, int lane) {
  const _Float16* col = p + (lane & 15) * ld + ((lane & 16) >> 1);
  h8 lo = *(const h8*)(col);
  h8 hi = *(const h8*)(col + 16);
  v16h r;
#pragma unroll
  for (int s = 0; s < 8; ++s) { r[s] = lo[s]; r[s + 8] = hi[s]; }
  return r;
}

// B fragment 32x16 from K-major f16 in LDS (V staging buffer)
__device__ __forceinline__ v16h b_frag_lds_f16(const _Float16* p, int ld, int lane) {
  const int n = lane & 15, kb = (lane & 16) >> 1;
  v16h r;
#pragma unroll
  for (int s = 0; s < 16; ++s) r[s] = p[(kb + s + ((s >= 8) ? 8 : 0)) * ld + n];
  return r;
}

// ---------------- CDNA5 async global->LDS copy (inline asm) -------------
__device__ __forceinline__ uint32_t lds_off(const void* p) {
  return (uint32_t)(uintptr_t)p;          // low 32 bits of generic LDS addr
}
__device__ __forceinline__ void async_copy_b128(uint32_t dst_lds, const void* gsrc) {
  asm volatile("global_load_async_to_lds_b128 %0, %1, off"
               :: "v"(dst_lds), "v"((unsigned long long)(uintptr_t)gsrc)
               : "memory");
}
__device__ __forceinline__ void wait_async_le2() {
  asm volatile("s_wait_asynccnt 0x2" ::: "memory");
}
__device__ __forceinline__ void wait_async_0() {
  asm volatile("s_wait_asynccnt 0x0" ::: "memory");
}

// ---------------- kernel 0: pack a 512x512 f32 weight into f16 N-major ---
__global__ void pack_wt(const float* __restrict__ W, _Float16* __restrict__ Wt) {
  int i = blockIdx.x * blockDim.x + threadIdx.x;        // over DM_*DM_
  if (i >= DM_ * DM_) return;
  int k = i >> 9, n = i & 511;                          // coalesced read
  Wt[(size_t)n * DM_ + k] = (_Float16)W[i];
}

// ---------------- kernel 1: pack E[1:] into head-split f16 ----------------
__global__ void pack_eh(const float* __restrict__ E, _Float16* __restrict__ Eh) {
  int i = blockIdx.x * blockDim.x + threadIdx.x;        // over L_*DM_
  if (i >= L_ * DM_) return;
  int k = i >> 9, c = i & 511;
  int h = c >> 6, d = c & 63;
  Eh[((size_t)h * L_ + k) * DEP_ + d] = (_Float16)E[DM_ + i];   // E row (1+k)
}

// ---------------- kernel 2: projection GEMM -> (B,H,L,64) f16 ------------
// Weights pre-packed f16 N-major: every B fragment = two 16B vector loads.
__global__ void gemm_proj(const float* __restrict__ X, const _Float16* __restrict__ Wt,
                          const float* __restrict__ bias, _Float16* __restrict__ Out) {
  const int lane = threadIdx.x & 31;
  const int wid  = blockIdx.x * 8 + (threadIdx.x >> 5);
  const int tm   = wid >> 5;                 // 240 M-tiles
  const int tn   = wid & 31;                 // 32 N-tiles
  const int m0 = tm * 16, n0 = tn * 16;
  const int bb = tm / (L_ / 16);             // tile never straddles a batch
  const int l0 = (tm % (L_ / 16)) * 16;
  v8f acc = {};
#pragma unroll 4
  for (int k0 = 0; k0 < DM_; k0 += 32) {
    __builtin_prefetch(X + (size_t)m0 * DM_ + k0 + 128, 0, 1);  // global_prefetch
    acc = wmma_f16(a_frag_f32(X + (size_t)m0 * DM_ + k0, DM_, lane),
                   b_frag_T_f16(Wt + (size_t)n0 * DM_ + k0, DM_, lane), acc);
  }
  const int mloc = (lane & 16) ? 8 : 0;
  const int n    = n0 + (lane & 15);
  const int h    = n >> 6, d = n & 63;
  const float bv = bias[n];
#pragma unroll
  for (int r = 0; r < 8; ++r)
    Out[(((size_t)bb * H_ + h) * L_ + l0 + mloc + r) * DEP_ + d] =
        (_Float16)(acc[r] + bv);
}

// ---------------- kernel 3: R[b,h,q,r] = qh . eh^T (band only) -----------
__global__ void relpos_gemm(const _Float16* __restrict__ Qh,
                            const _Float16* __restrict__ Eh,
                            _Float16* __restrict__ R) {
  const int lane = threadIdx.x & 31;
  const int wid  = blockIdx.x * 8 + (threadIdx.x >> 5);
  const int bh   = wid / (LT_ * LT_);
  const int rest = wid % (LT_ * LT_);
  const int qt = rest / LT_, rt = rest % LT_;
  const int q0 = qt * 16, r0 = rt * 16;
  // causal gather touches only r >= L_-1-q : skip dead tiles (wave-uniform)
  if (q0 + r0 < (L_ - 31)) return;
  const int h = bh & 7;
  const _Float16* Ab = Qh + ((size_t)bh * L_ + q0) * DEP_;
  const _Float16* Bb = Eh + ((size_t)h  * L_ + r0) * DEP_;
  v8f acc = {};
#pragma unroll
  for (int d0 = 0; d0 < DEP_; d0 += 32)
    acc = wmma_f16(a_frag_f16(Ab + d0, DEP_, lane),
                   b_frag_T_f16(Bb + d0, DEP_, lane), acc);
  const int mloc = (lane & 16) ? 8 : 0;
  const int n    = lane & 15;
#pragma unroll
  for (int r = 0; r < 8; ++r)
    R[((size_t)bh * L_ + q0 + mloc + r) * L_ + r0 + n] = (_Float16)acc[r];
}

// ---------------- kernel 4: fused attention per (b,h,q-tile) -------------
// 128 threads / 4 waves.  Dynamic LDS: 16 x L_ f32 logits strip (122,880 B).
// Static LDS: double-buffered 32x64 f16 V tile filled by async copies.
__global__ void attn_kernel(const _Float16* __restrict__ Qh,
                            const _Float16* __restrict__ Kh,
                            const _Float16* __restrict__ Vh,
                            const _Float16* __restrict__ R,
                            _Float16* __restrict__ Ctx,
                            float* __restrict__ AttnOut) {
  extern __shared__ float sm[];                 // [16][L_]
  __shared__ _Float16 vbuf[2][32 * DEP_];       // 2 x 4 KB V staging
  __shared__ float red[16 * 8];
  __shared__ float rowmax[16], rowinv[16];
  const int tid = threadIdx.x, lane = tid & 31, wave = tid >> 5;
  const int qt = blockIdx.x % LT_;
  const int bh = blockIdx.x / LT_;
  const int q0 = qt * 16;
  const size_t base = (size_t)bh * L_;
  const int valid = (qt + 1) * 16;              // causal column count

  // kick off async fetch of V chunk 0 immediately (overlaps passes 1-2)
  {
    const _Float16* g = Vh + base * DEP_ + tid * 16;
    const uint32_t  l = lds_off(&vbuf[0][tid * 16]);
    async_copy_b128(l,      g);
    async_copy_b128(l + 16, g + 8);
  }

  // ---- pass 1: logits = (QK^T + Srel)/8 into LDS (causal tiles only) ----
  const v16h aq0 = a_frag_f16(Qh + (base + q0) * DEP_,      DEP_, lane);
  const v16h aq1 = a_frag_f16(Qh + (base + q0) * DEP_ + 32, DEP_, lane);
  for (int kt = wave; kt <= qt; kt += 4) {
    const int k0 = kt * 16;
    v8f acc = {};
    acc = wmma_f16(aq0, b_frag_T_f16(Kh + (base + k0) * DEP_,      DEP_, lane), acc);
    acc = wmma_f16(aq1, b_frag_T_f16(Kh + (base + k0) * DEP_ + 32, DEP_, lane), acc);
    const int mloc = (lane & 16) ? 8 : 0;
    const int gk   = k0 + (lane & 15);
    if (kt < qt) {                               // fully unmasked tile
#pragma unroll
      for (int r = 0; r < 8; ++r) {
        const int m  = mloc + r;
        const int gq = q0 + m;
        const float srel = (float)R[(base + gq) * L_ + (gk - gq + (L_ - 1))];
        sm[m * L_ + gk] = (acc[r] + srel) * 0.125f;
      }
    } else {                                     // diagonal tile: ragged mask
#pragma unroll
      for (int r = 0; r < 8; ++r) {
        const int m  = mloc + r;
        const int gq = q0 + m;
        float v = -1.0e30f;
        if (gk <= gq)
          v = (acc[r] + (float)R[(base + gq) * L_ + (gk - gq + (L_ - 1))]) * 0.125f;
        sm[m * L_ + gk] = v;
      }
    }
  }
  __syncthreads();

  // ---- rowwise max over the valid prefix (8 partials per row) ----
  {
    const int row = tid >> 3, sub = tid & 7;
    const int chunk = valid >> 3;                // valid % 8 == 0
    const float* rp = sm + row * L_ + sub * chunk;
    float pm = -3.0e38f;
    for (int i = 0; i < chunk; ++i) pm = fmaxf(pm, rp[i]);
    red[row * 8 + sub] = pm;
  }
  __syncthreads();
  if (tid < 16) {
    float pm = red[tid * 8];
#pragma unroll
    for (int i = 1; i < 8; ++i) pm = fmaxf(pm, red[tid * 8 + i]);
    rowmax[tid] = pm;
  }
  __syncthreads();
  // ---- rowwise exp-sum; store e = exp(l - max) back in place ----
  {
    const int row = tid >> 3, sub = tid & 7;
    const int chunk = valid >> 3;
    const float mx = rowmax[row];
    float* rp = sm + row * L_ + sub * chunk;
    float s = 0.f;
    for (int i = 0; i < chunk; ++i) {
      const float e = __expf(rp[i] - mx);
      rp[i] = e;
      s += e;
    }
    red[row * 8 + sub] = s;
  }
  __syncthreads();
  if (tid < 16) {
    float s = 0.f;
#pragma unroll
    for (int i = 0; i < 8; ++i) s += red[tid * 8 + i];
    rowinv[tid] = 1.0f / s;
  }
  __syncthreads();

  // ---- pass 2: scale stored e by 1/sum + stream attn; zero masked tail ---
  float* aout = AttnOut + (base + q0) * L_;
#pragma unroll 1
  for (int m = 0; m < 16; ++m) {
    const float inv = rowinv[m];
    for (int c = tid; c < valid; c += 128) {
      const float p = sm[m * L_ + c] * inv;
      sm[m * L_ + c]   = p;
      aout[m * L_ + c] = p;
    }
    for (int c = valid + tid; c < L_; c += 128) {
      sm[m * L_ + c]   = 0.f;
      aout[m * L_ + c] = 0.f;
    }
  }
  __syncthreads();

  // ---- pass 3: ctx = probs @ V, V double-buffered via async LDS copies ---
  const int n0 = wave * 16;                     // wave owns depth cols
  const int kend = (valid + 31) & ~31;
  const int nchunks = kend >> 5;
  v8f acc = {};
  for (int ci = 0; ci < nchunks; ++ci) {
    if (ci + 1 < nchunks) {                     // prefetch next 32 V rows
      const _Float16* g = Vh + (base + (size_t)(ci + 1) * 32) * DEP_ + tid * 16;
      const uint32_t  l = lds_off(&vbuf[(ci + 1) & 1][tid * 16]);
      async_copy_b128(l,      g);
      async_copy_b128(l + 16, g + 8);
      wait_async_le2();                         // current chunk complete
    } else {
      wait_async_0();
    }
    __syncthreads();                            // all waves' copies visible
    acc = wmma_f16(a_frag_lds(sm, L_, ci * 32, lane),
                   b_frag_lds_f16(&vbuf[ci & 1][n0], DEP_, lane), acc);
    __syncthreads();                            // done reading before refill
  }
  const int mloc = (lane & 16) ? 8 : 0, n = lane & 15;
  const int bb = bh >> 3, h = bh & 7;
#pragma unroll
  for (int r = 0; r < 8; ++r)
    Ctx[((size_t)bb * L_ + q0 + mloc + r) * DM_ + h * DEP_ + n0 + n] =
        (_Float16)acc[r];
}

// ---------------- kernel 5: out = ctx @ Wo + bo (f32) --------------------
__global__ void gemm_out_k(const _Float16* __restrict__ Ctx, const _Float16* __restrict__ Wot,
                           const float* __restrict__ bo, float* __restrict__ Out) {
  const int lane = threadIdx.x & 31;
  const int wid  = blockIdx.x * 8 + (threadIdx.x >> 5);
  const int tm = wid >> 5, tn = wid & 31;
  const int m0 = tm * 16, n0 = tn * 16;
  v8f acc = {};
#pragma unroll 4
  for (int k0 = 0; k0 < DM_; k0 += 32) {
    acc = wmma_f16(a_frag_f16(Ctx + (size_t)m0 * DM_ + k0, DM_, lane),
                   b_frag_T_f16(Wot + (size_t)n0 * DM_ + k0, DM_, lane), acc);
  }
  const int mloc = (lane & 16) ? 8 : 0;
  const int n    = n0 + (lane & 15);
  const float bv = bo[n];
#pragma unroll
  for (int r = 0; r < 8; ++r)
    Out[(size_t)(m0 + mloc + r) * DM_ + n] = acc[r] + bv;
}

// ---------------- host launch ----------------
extern "C" void kernel_launch(void* const* d_in, const int* in_sizes, int n_in,
                              void* d_out, int out_size, void* d_ws, size_t ws_size,
                              hipStream_t stream) {
  (void)in_sizes; (void)n_in; (void)out_size; (void)ws_size;
  const float* q  = (const float*)d_in[0];
  const float* k  = (const float*)d_in[1];
  const float* v  = (const float*)d_in[2];
  /* d_in[3] (mask) unused: causal mask applied analytically */
  const float* Wq = (const float*)d_in[4];
  const float* bq = (const float*)d_in[5];
  const float* Wk = (const float*)d_in[6];
  const float* bk = (const float*)d_in[7];
  const float* Wv = (const float*)d_in[8];
  const float* bv = (const float*)d_in[9];
  const float* E  = (const float*)d_in[10];
  const float* Wo = (const float*)d_in[11];
  const float* bo = (const float*)d_in[12];

  char* ws = (char*)d_ws;
  size_t off = 0;
  auto carve = [&](size_t bytes) {
    void* p = ws + off;
    off = (off + bytes + 255) & ~(size_t)255;
    return p;
  };
  const size_t BHLD = (size_t)BH_ * L_ * DEP_;
  _Float16* Qh  = (_Float16*)carve(BHLD * 2);
  _Float16* Kh  = (_Float16*)carve(BHLD * 2);
  _Float16* Vh  = (_Float16*)carve(BHLD * 2);
  _Float16* Eh  = (_Float16*)carve((size_t)H_ * L_ * DEP_ * 2);
  _Float16* Ctx = (_Float16*)carve((size_t)M_ * DM_ * 2);
  _Float16* Wqt = (_Float16*)carve((size_t)DM_ * DM_ * 2);
  _Float16* Wkt = (_Float16*)carve((size_t)DM_ * DM_ * 2);
  _Float16* Wvt = (_Float16*)carve((size_t)DM_ * DM_ * 2);
  _Float16* Wot = (_Float16*)carve((size_t)DM_ * DM_ * 2);
  _Float16* R   = (_Float16*)carve((size_t)BH_ * L_ * L_ * 2);   // ~118 MB

  float* Out  = (float*)d_out;
  float* Attn = Out + (size_t)B_ * L_ * DM_;

  // weight / embedding packing (tiny, L2-resident afterwards)
  pack_wt<<<(DM_ * DM_ + 255) / 256, 256, 0, stream>>>(Wq, Wqt);
  pack_wt<<<(DM_ * DM_ + 255) / 256, 256, 0, stream>>>(Wk, Wkt);
  pack_wt<<<(DM_ * DM_ + 255) / 256, 256, 0, stream>>>(Wv, Wvt);
  pack_wt<<<(DM_ * DM_ + 255) / 256, 256, 0, stream>>>(Wo, Wot);
  pack_eh<<<(L_ * DM_ + 255) / 256, 256, 0, stream>>>(E, Eh);

  gemm_proj<<<(M_ / 16) * (DM_ / 16) / 8, 256, 0, stream>>>(q, Wqt, bq, Qh);
  gemm_proj<<<(M_ / 16) * (DM_ / 16) / 8, 256, 0, stream>>>(k, Wkt, bk, Kh);
  gemm_proj<<<(M_ / 16) * (DM_ / 16) / 8, 256, 0, stream>>>(v, Wvt, bv, Vh);

  relpos_gemm<<<(BH_ * LT_ * LT_) / 8, 256, 0, stream>>>(Qh, Eh, R);

  const size_t smem = (size_t)16 * L_ * sizeof(float);   // 122,880 B
  attn_kernel<<<BH_ * LT_, 128, smem, stream>>>(Qh, Kh, Vh, R, Ctx, Attn);

  gemm_out_k<<<(M_ / 16) * (DM_ / 16) / 8, 256, 0, stream>>>(Ctx, Wot, bo, Out);
}
